// Decoder_50148038148527
// MI455X (gfx1250) — matrix operand order repaired and checked by reading, not compiled, not run on hardware
//
#include <hip/hip_runtime.h>
#include <hip/hip_bf16.h>

// ---------------------------------------------------------------------------
// CDNA5 (gfx1250) wave32 WMMA decoder forward pass.
// All GEMMs + attention use v_wmma_f32_16x16x32_bf16 (fp32 inputs converted to
// bf16 at LDS-staging time, fp32 accumulation). LDS layouts are arranged so
// every WMMA fragment is two contiguous 16B runs per lane -> ds_load_b128.
// Epilogue flags are compile-time template parameters (no runtime branches).
// ---------------------------------------------------------------------------

typedef __attribute__((ext_vector_type(16))) __bf16 v16bf;
typedef __attribute__((ext_vector_type(8)))  float  v8f;

#define DEC_B 2
#define DEC_S 2048
#define DEC_D 1024
#define DEC_H 16
#define DEC_DH 64
#define DEC_F 4096
#define DEC_V 32000
#define DEC_L 2
#define DEC_M (DEC_B * DEC_S)   // 4096 token rows

static __device__ __forceinline__ __bf16 f2bf(float f) {
  unsigned u = __builtin_bit_cast(unsigned, f);
  u += 0x7FFFu + ((u >> 16) & 1u);          // round-to-nearest-even
  unsigned short h = (unsigned short)(u >> 16);
  return __builtin_bit_cast(__bf16, h);
}

// A fragment: 16x32 bf16, lane = M (0..15 twice), K pairs per ISA 7.12.2.
// at(m, kk) must be contiguous in kk per lane for b128 merging.
template <typename AT>
static __device__ __forceinline__ v16bf load_a_frag(int lane, AT at) {
  const int m  = lane & 15;
  const int kb = (lane < 16) ? 0 : 8;
  v16bf f;
#pragma unroll
  for (int j = 0; j < 8; ++j) {
    const int kk = (j < 4) ? (kb + 2 * j) : (kb + 16 + 2 * (j - 4));
    f[2 * j]     = at(m, kk);
    f[2 * j + 1] = at(m, kk + 1);
  }
  return f;
}

// B fragment: 32x16 bf16, lane = N, K pairs per ISA 7.12.2 (same order as A).
// bt(kk, n) must be contiguous in kk per lane for b128 merging (use
// K-major/transposed LDS tiles).
template <typename BT>
static __device__ __forceinline__ v16bf load_b_frag(int lane, BT bt) {
  const int n  = lane & 15;
  const int kb = (lane < 16) ? 0 : 8;
  v16bf f;
#pragma unroll
  for (int j = 0; j < 8; ++j) {
    const int kk = (j < 4) ? (kb + 2 * j) : (kb + 16 + 2 * (j - 4));
    f[2 * j]     = bt(kk, n);
    f[2 * j + 1] = bt(kk + 1, n);
  }
  return f;
}

// ---------------------------------------------------------------------------
// Embedding gather:  x[b*S+s, :] = emb[tokens[b,s], :] + pe[s, :]
// ---------------------------------------------------------------------------
__global__ __launch_bounds__(256) void embed_kernel(
    const int* __restrict__ tokens, const float* __restrict__ emb,
    const float* __restrict__ pe, float* __restrict__ x) {
  const int row = blockIdx.x;               // b*S + s
  const int s   = row % DEC_S;
  const int tok = tokens[row];
  const float* er = emb + (size_t)tok * DEC_D;
  const float* pr = pe  + (size_t)s   * DEC_D;
  float* xr = x + (size_t)row * DEC_D;
#pragma unroll
  for (int i = 0; i < DEC_D / 256; ++i) {
    const int d = threadIdx.x + i * 256;
    xr[d] = er[d] + pr[d];
  }
}

// ---------------------------------------------------------------------------
// LayerNorm over D=1024, one block per row.
// ---------------------------------------------------------------------------
__global__ __launch_bounds__(256) void ln_kernel(
    const float* __restrict__ x, const float* __restrict__ g,
    const float* __restrict__ b, float* __restrict__ out) {
  __shared__ float red[256];
  const int row = blockIdx.x;
  const float* xr = x + (size_t)row * DEC_D;
  float v[DEC_D / 256];
  float s = 0.f;
#pragma unroll
  for (int i = 0; i < DEC_D / 256; ++i) {
    v[i] = xr[threadIdx.x + i * 256];
    s += v[i];
  }
  red[threadIdx.x] = s;
  __syncthreads();
  for (int off = 128; off > 0; off >>= 1) {
    if (threadIdx.x < off) red[threadIdx.x] += red[threadIdx.x + off];
    __syncthreads();
  }
  const float mean = red[0] * (1.0f / DEC_D);
  __syncthreads();
  s = 0.f;
#pragma unroll
  for (int i = 0; i < DEC_D / 256; ++i) {
    const float d = v[i] - mean;
    s += d * d;
  }
  red[threadIdx.x] = s;
  __syncthreads();
  for (int off = 128; off > 0; off >>= 1) {
    if (threadIdx.x < off) red[threadIdx.x] += red[threadIdx.x + off];
    __syncthreads();
  }
  const float rstd = rsqrtf(red[0] * (1.0f / DEC_D) + 1e-4f);
  float* orow = out + (size_t)row * DEC_D;
#pragma unroll
  for (int i = 0; i < DEC_D / 256; ++i) {
    const int d = threadIdx.x + i * 256;
    orow[d] = (v[i] - mean) * rstd * g[d] + b[d];
  }
}

// ---------------------------------------------------------------------------
// Generic bf16-WMMA GEMM:  C[M,N] = A[M,K] @ B[K,N]  (+bias, +residual, relu)
// fp32 in memory -> bf16 in LDS. B staged TRANSPOSED (BsT[n][k]) so B-frag
// loads are contiguous per lane. Block tile 128x128, K-tile 64, 8 waves
// (4m x 2n), each wave 32x64 -> 16 WMMAs per K-tile.
// BIAS / RELU / RES are compile-time: branch-free vectorizable epilogue.
// ---------------------------------------------------------------------------
#define GAS 72    // As  row stride (elements): 64 K + 8 pad  (144B, 16B mult)
#define GBT 72    // BsT row stride (elements): 64 K + 8 pad

template <int BIAS, int RELU, int RES>
__global__ __launch_bounds__(256) void gemm_bf16_kernel(
    const float* __restrict__ A, const float* __restrict__ B,
    const float* __restrict__ bias, const float* __restrict__ resid,
    float* __restrict__ C, int M, int N, int K) {
  __shared__ __bf16 As [128 * GAS];   // [m][k]  18KB
  __shared__ __bf16 BsT[128 * GBT];   // [n][k]  18KB

  const int tid  = threadIdx.x;
  const int lane = tid & 31;
  const int wid  = tid >> 5;
  const int wm   = (wid & 3) * 32;   // wave M offset in tile
  const int wn   = (wid >> 2) * 64;  // wave N offset in tile
  const int mBase = blockIdx.y * 128;
  const int nBase = blockIdx.x * 128;

  v8f acc[2][4];
#pragma unroll
  for (int mi = 0; mi < 2; ++mi)
#pragma unroll
    for (int nf = 0; nf < 4; ++nf)
#pragma unroll
      for (int e = 0; e < 8; ++e) acc[mi][nf][e] = 0.f;

  for (int k0 = 0; k0 < K; k0 += 64) {
    __syncthreads();
    // Stage A tile (128 x 64): 2048 float4 loads, row-major bf16
#pragma unroll
    for (int i = 0; i < 8; ++i) {
      const int vid = tid + i * 256;
      const int r   = vid >> 4;
      const int kk  = (vid & 15) * 4;
      const float4 a4 = *(const float4*)(A + (size_t)(mBase + r) * K + k0 + kk);
      As[r * GAS + kk + 0] = f2bf(a4.x);
      As[r * GAS + kk + 1] = f2bf(a4.y);
      As[r * GAS + kk + 2] = f2bf(a4.z);
      As[r * GAS + kk + 3] = f2bf(a4.w);
    }
    // Stage B tile (64 x 128) transposed into BsT[n][k]
#pragma unroll
    for (int i = 0; i < 8; ++i) {
      const int vid = tid + i * 256;
      const int kr  = vid >> 5;            // 0..63
      const int nn  = (vid & 31) * 4;      // 0..124
      const float4 b4 = *(const float4*)(B + (size_t)(k0 + kr) * N + nBase + nn);
      BsT[(nn + 0) * GBT + kr] = f2bf(b4.x);
      BsT[(nn + 1) * GBT + kr] = f2bf(b4.y);
      BsT[(nn + 2) * GBT + kr] = f2bf(b4.z);
      BsT[(nn + 3) * GBT + kr] = f2bf(b4.w);
    }
    __syncthreads();

#pragma unroll
    for (int ks = 0; ks < 64; ks += 32) {
      // Load ALL fragments first, then an 8-WMMA burst (DS loads overlap,
      // WMMAs pipeline back-to-back).
      v16bf af[2];
#pragma unroll
      for (int mi = 0; mi < 2; ++mi)
        af[mi] = load_a_frag(lane, [&](int m, int kk) -> __bf16 {
          return As[(wm + mi * 16 + m) * GAS + ks + kk];
        });
      v16bf bf_[4];
#pragma unroll
      for (int nf = 0; nf < 4; ++nf)
        bf_[nf] = load_b_frag(lane, [&](int kk, int n) -> __bf16 {
          return BsT[(wn + nf * 16 + n) * GBT + ks + kk];
        });
#pragma unroll
      for (int nf = 0; nf < 4; ++nf)
#pragma unroll
        for (int mi = 0; mi < 2; ++mi)
          acc[mi][nf] = __builtin_amdgcn_wmma_f32_16x16x32_bf16(
              false, af[mi], false, bf_[nf], (short)0, acc[mi][nf], false, false);
    }
  }

  // Epilogue: C layout lane = N, VGPR r = M (lanes 16..31 -> M += 8)
  const int half8 = (lane < 16) ? 0 : 8;
  const int nlo   = lane & 15;
  float bias_v[4];
#pragma unroll
  for (int nf = 0; nf < 4; ++nf)
    bias_v[nf] = BIAS ? bias[nBase + wn + nf * 16 + nlo] : 0.f;
#pragma unroll
  for (int mi = 0; mi < 2; ++mi) {
#pragma unroll
    for (int nf = 0; nf < 4; ++nf) {
#pragma unroll
      for (int r = 0; r < 8; ++r) {
        const int row = mBase + wm + mi * 16 + r + half8;
        const int col = nBase + wn + nf * 16 + nlo;
        float val = acc[mi][nf][r];
        if (BIAS) val += bias_v[nf];
        if (RELU) val = fmaxf(val, 0.f);
        if (RES)  val += resid[(size_t)row * N + col];
        C[(size_t)row * N + col] = val;
      }
    }
  }
}

// ---------------------------------------------------------------------------
// Flash attention: one block = (b, h, 64 query rows). 4 waves x 16 q-rows.
// QK^T and P@V via v_wmma_f32_16x16x32_bf16, online softmax in registers.
// V staged transposed (VsT[d][key]) so P@V B-frags are contiguous per lane.
// ---------------------------------------------------------------------------
#define FAS 72   // LDS row stride for 64-wide bf16 tiles (144B)

__global__ __launch_bounds__(128) void attn_kernel(
    const float* __restrict__ q, const float* __restrict__ k,
    const float* __restrict__ v, float* __restrict__ ctx) {
  __shared__ __bf16 Qs [64 * FAS];    // [q-row][d]
  __shared__ __bf16 Ks [64 * FAS];    // [key][d]   (QK^T B-frag: contiguous d)
  __shared__ __bf16 VsT[64 * FAS];    // [d][key]   (PV  B-frag: contiguous key)
  __shared__ __bf16 Ps[4][16 * FAS];  // per-wave P tile [q-row][key]

  const int tid  = threadIdx.x;
  const int lane = tid & 31;
  const int wid  = tid >> 5;
  const int bh   = blockIdx.y;
  const int b    = bh >> 4;
  const int h    = bh & 15;
  const int q0   = blockIdx.x * 64;
  const int rowBase  = b * DEC_S;
  const int headBase = h * DEC_DH;
  const float scale  = 0.125f;   // 1/sqrt(64)

  // Stage Q tile (64 x 64 head slice), fp32 -> bf16
#pragma unroll
  for (int i = 0; i < 8; ++i) {
    const int vid = tid + i * 128;
    const int r   = vid >> 4;
    const int c   = (vid & 15) * 4;
    const float4 q4 =
        *(const float4*)(q + (size_t)(rowBase + q0 + r) * DEC_D + headBase + c);
    Qs[r * FAS + c + 0] = f2bf(q4.x);
    Qs[r * FAS + c + 1] = f2bf(q4.y);
    Qs[r * FAS + c + 2] = f2bf(q4.z);
    Qs[r * FAS + c + 3] = f2bf(q4.w);
  }
  __syncthreads();

  // Q fragments for this wave's 16 rows (K = Dh = 64 -> two 32-wide chunks)
  v16bf aq[2];
#pragma unroll
  for (int kk = 0; kk < 2; ++kk)
    aq[kk] = load_a_frag(lane, [&](int m, int kc) -> __bf16 {
      return Qs[(wid * 16 + m) * FAS + kk * 32 + kc];
    });

  v8f O[4];
#pragma unroll
  for (int df = 0; df < 4; ++df)
#pragma unroll
    for (int e = 0; e < 8; ++e) O[df][e] = 0.f;
  float row_m[8], row_l[8];
#pragma unroll
  for (int r = 0; r < 8; ++r) { row_m[r] = -1e30f; row_l[r] = 0.f; }

  const int half8 = (lane < 16) ? 0 : 8;
  const int nlo   = lane & 15;
  __bf16* Ps_w = &Ps[wid][0];

  for (int kt = 0; kt <= blockIdx.x; ++kt) {
    const int k0 = kt * 64;
    __syncthreads();
    // Stage K (row-major) and V (transposed) tiles, each 64 x 64
#pragma unroll
    for (int i = 0; i < 8; ++i) {
      const int vid = tid + i * 128;
      const int r   = vid >> 4;
      const int c   = (vid & 15) * 4;
      const float4 k4 =
          *(const float4*)(k + (size_t)(rowBase + k0 + r) * DEC_D + headBase + c);
      Ks[r * FAS + c + 0] = f2bf(k4.x);
      Ks[r * FAS + c + 1] = f2bf(k4.y);
      Ks[r * FAS + c + 2] = f2bf(k4.z);
      Ks[r * FAS + c + 3] = f2bf(k4.w);
      const float4 v4 =
          *(const float4*)(v + (size_t)(rowBase + k0 + r) * DEC_D + headBase + c);
      VsT[(c + 0) * FAS + r] = f2bf(v4.x);
      VsT[(c + 1) * FAS + r] = f2bf(v4.y);
      VsT[(c + 2) * FAS + r] = f2bf(v4.z);
      VsT[(c + 3) * FAS + r] = f2bf(v4.w);
    }
    __syncthreads();

    // S = Q @ K^T  -> 16 q-rows x 64 keys per wave
    v16bf bk[4];
#pragma unroll
    for (int nf = 0; nf < 4; ++nf)
      bk[nf] = load_b_frag(lane, [&](int kc, int n) -> __bf16 {
        return Ks[(nf * 16 + n) * FAS + kc];          // chunk kk=0
      });
    v8f sacc[4];
#pragma unroll
    for (int nf = 0; nf < 4; ++nf)
#pragma unroll
      for (int e = 0; e < 8; ++e) sacc[nf][e] = 0.f;
#pragma unroll
    for (int nf = 0; nf < 4; ++nf)
      sacc[nf] = __builtin_amdgcn_wmma_f32_16x16x32_bf16(
          false, aq[0], false, bk[nf], (short)0, sacc[nf], false, false);
#pragma unroll
    for (int nf = 0; nf < 4; ++nf)
      bk[nf] = load_b_frag(lane, [&](int kc, int n) -> __bf16 {
        return Ks[(nf * 16 + n) * FAS + 32 + kc];     // chunk kk=1
      });
#pragma unroll
    for (int nf = 0; nf < 4; ++nf)
      sacc[nf] = __builtin_amdgcn_wmma_f32_16x16x32_bf16(
          false, aq[1], false, bk[nf], (short)0, sacc[nf], false, false);

    // Online softmax. Row m of the 16x16 C-tile = VGPR index (+8 for lanes>=16).
#pragma unroll
    for (int r = 0; r < 8; ++r) {
      const int qi = q0 + wid * 16 + half8 + r;
      float mx = -1e30f;
#pragma unroll
      for (int nf = 0; nf < 4; ++nf) {
        float val = sacc[nf][r] * scale;
        const int kj = k0 + nf * 16 + nlo;
        if (kj > qi) val = -1e30f;      // causal mask
        sacc[nf][r] = val;
        mx = fmaxf(mx, val);
      }
#pragma unroll
      for (int msk = 1; msk < 16; msk <<= 1)
        mx = fmaxf(mx, __shfl_xor(mx, msk, 32));
      const float mnew  = fmaxf(row_m[r], mx);
      const float alpha = __expf(row_m[r] - mnew);
      float lsum = 0.f;
#pragma unroll
      for (int nf = 0; nf < 4; ++nf) {
        const float pv = __expf(sacc[nf][r] - mnew);
        Ps_w[(half8 + r) * FAS + nf * 16 + nlo] = f2bf(pv);
        lsum += pv;
      }
#pragma unroll
      for (int msk = 1; msk < 16; msk <<= 1)
        lsum += __shfl_xor(lsum, msk, 32);
      row_l[r] = row_l[r] * alpha + lsum;
      row_m[r] = mnew;
#pragma unroll
      for (int df = 0; df < 4; ++df) O[df][r] *= alpha;
    }
    __builtin_amdgcn_wave_barrier();   // keep P stores before P loads

    // O += P @ V   (keys = K dim, two 32-wide chunks)
#pragma unroll
    for (int kk2 = 0; kk2 < 2; ++kk2) {
      const v16bf ap = load_a_frag(lane, [&](int m, int kc) -> __bf16 {
        return Ps_w[m * FAS + kk2 * 32 + kc];
      });
      v16bf bv[4];
#pragma unroll
      for (int df = 0; df < 4; ++df)
        bv[df] = load_b_frag(lane, [&](int kc, int n) -> __bf16 {
          return VsT[(df * 16 + n) * FAS + kk2 * 32 + kc];  // B[key][d]=VsT[d][key]
        });
#pragma unroll
      for (int df = 0; df < 4; ++df)
        O[df] = __builtin_amdgcn_wmma_f32_16x16x32_bf16(
            false, ap, false, bv[df], (short)0, O[df], false, false);
    }
  }

  // Normalize and write ctx
#pragma unroll
  for (int df = 0; df < 4; ++df) {
#pragma unroll
    for (int r = 0; r < 8; ++r) {
      const int row = rowBase + q0 + wid * 16 + half8 + r;
      const int col = headBase + df * 16 + nlo;
      ctx[(size_t)row * DEC_D + col] = O[df][r] / row_l[r];
    }
  }
}

// ---------------------------------------------------------------------------
// Host-side orchestration
// ---------------------------------------------------------------------------
extern "C" void kernel_launch(void* const* d_in, const int* in_sizes, int n_in,
                              void* d_out, int out_size, void* d_ws, size_t ws_size,
                              hipStream_t stream) {
  (void)in_sizes; (void)n_in; (void)out_size; (void)ws_size;
  const int*   tokens = (const int*)  d_in[0];
  const float* emb    = (const float*)d_in[1];
  const float* pe     = (const float*)d_in[2];
  const float* ln1_g  = (const float*)d_in[3];
  const float* ln1_b  = (const float*)d_in[4];
  const float* Wq     = (const float*)d_in[5];
  const float* Wk     = (const float*)d_in[6];
  const float* Wv     = (const float*)d_in[7];
  const float* Wo     = (const float*)d_in[8];
  const float* bo     = (const float*)d_in[9];
  const float* ln2_g  = (const float*)d_in[10];
  const float* ln2_b  = (const float*)d_in[11];
  const float* W1     = (const float*)d_in[12];
  const float* b1     = (const float*)d_in[13];
  const float* W2     = (const float*)d_in[14];
  const float* b2     = (const float*)d_in[15];
  const float* lnf_g  = (const float*)d_in[16];
  const float* lnf_b  = (const float*)d_in[17];
  const float* Wh     = (const float*)d_in[18];
  const float* bh     = (const float*)d_in[19];
  float* logits = (float*)d_out;

  const int M = DEC_M, D = DEC_D, F = DEC_F, V = DEC_V;

  float* x   = (float*)d_ws;         // [M,D] residual stream
  float* hbuf= x    + (size_t)M * D; // [M,D] LN output
  float* qb  = hbuf + (size_t)M * D; // [M,D]
  float* kb  = qb   + (size_t)M * D; // [M,D]
  float* vb  = kb   + (size_t)M * D; // [M,D]
  float* ctx = vb   + (size_t)M * D; // [M,D]
  float* ff  = ctx  + (size_t)M * D; // [M,F]

  embed_kernel<<<M, 256, 0, stream>>>(tokens, emb, pe, x);

  for (int l = 0; l < DEC_L; ++l) {
    const float* wq = Wq + (size_t)l * D * D;
    const float* wk = Wk + (size_t)l * D * D;
    const float* wv = Wv + (size_t)l * D * D;
    const float* wo = Wo + (size_t)l * D * D;
    const float* w1 = W1 + (size_t)l * D * F;
    const float* w2 = W2 + (size_t)l * F * D;

    ln_kernel<<<M, 256, 0, stream>>>(x, ln1_g + l * D, ln1_b + l * D, hbuf);

    dim3 gDD(D / 128, M / 128);
    gemm_bf16_kernel<0,0,0><<<gDD, 256, 0, stream>>>(hbuf, wq, nullptr, nullptr, qb, M, D, D);
    gemm_bf16_kernel<0,0,0><<<gDD, 256, 0, stream>>>(hbuf, wk, nullptr, nullptr, kb, M, D, D);
    gemm_bf16_kernel<0,0,0><<<gDD, 256, 0, stream>>>(hbuf, wv, nullptr, nullptr, vb, M, D, D);

    dim3 gAttn(DEC_S / 64, DEC_B * DEC_H);
    attn_kernel<<<gAttn, 128, 0, stream>>>(qb, kb, vb, ctx);

    // x = ctx @ Wo + bo + x
    gemm_bf16_kernel<1,0,1><<<gDD, 256, 0, stream>>>(ctx, wo, bo + l * D, x, x, M, D, D);

    ln_kernel<<<M, 256, 0, stream>>>(x, ln2_g + l * D, ln2_b + l * D, hbuf);

    // ff = relu(h @ W1 + b1)
    dim3 gDF(F / 128, M / 128);
    gemm_bf16_kernel<1,1,0><<<gDF, 256, 0, stream>>>(hbuf, w1, b1 + l * F, nullptr, ff, M, F, D);
    // x = ff @ W2 + b2 + x
    gemm_bf16_kernel<1,0,1><<<gDD, 256, 0, stream>>>(ff, w2, b2 + l * D, x, x, M, D, F);
  }

  ln_kernel<<<M, 256, 0, stream>>>(x, lnf_g, lnf_b, hbuf);

  dim3 gHead(V / 128, M / 128);
  gemm_bf16_kernel<1,0,0><<<gHead, 256, 0, stream>>>(hbuf, Wh, bh, nullptr, logits, M, V, D);
}
